// CustomLoss_91113436218061
// MI455X (gfx1250) — matrix operand order repaired and checked by reading, not compiled, not run on hardware
//
#include <hip/hip_runtime.h>

typedef __attribute__((ext_vector_type(16))) _Float16 v16h;
typedef __attribute__((ext_vector_type(8)))  _Float16 v8h;
typedef __attribute__((ext_vector_type(8)))  float    v8f;

#define B_N   4096
#define D_N   512
#define NNEG  100
#define NDOM  8
#define TAUF  1e-4f
#define EPSF  1e-6f
#define REGW  1e-4f

#define NCG        16                   // column-tile groups for supp kernel
#define CT_PER_CG  (B_N / 16 / NCG)     // 16 column tiles per group

// acc slots: 0=mse_sq 1=sq(W_e2) 2=sq(b_e2) 3=sq(W_head) 4=sq(b_head) 5=sum L_a 6=sum L_s

// ---------- WMMA fragment loaders (row-major f16, row stride ld elements) ----------
// A: 16x32 (MxK). lane<16: M=lane, K = 0..15; lane>=16: M=lane-16, K = 16..31
__device__ inline v16h load_fragA(const _Float16* M, int ld, int row0, int k0) {
    int lane = threadIdx.x & 31;
    int r  = row0 + (lane & 15);
    int kb = (lane >> 4) * 16;
    const _Float16* p = M + r * ld + k0 + kb;
    v8h lo = *(const v8h*)(p);
    v8h hi = *(const v8h*)(p + 8);
    v16h f;
#pragma unroll
    for (int i = 0; i < 8; ++i) { f[i] = lo[i]; f[8 + i] = hi[i]; }
    return f;
}
// B: 32x16 (KxN), sourced row-major: B[k][n] = src[row0+n][k0+k].
// lane<16: N=lane, K=0..15; lane>=16: N=lane-16, K=16..31
__device__ inline v16h load_fragB(const _Float16* M, int ld, int row0, int k0) {
    int lane = threadIdx.x & 31;
    int r  = row0 + (lane & 15);
    int kb = (lane >> 4) * 16;
    return *(const v16h*)(M + r * ld + k0 + kb);
}

// ---------- small helper kernels ----------
__global__ void init_ws_kernel(float* acc, int* domCnt) {
    int t = threadIdx.x;
    if (t < 8) { acc[t] = 0.0f; domCnt[t] = 0; }
}

__global__ void zero_f32_kernel(float* p, int n) {
    int i = blockIdx.x * blockDim.x + threadIdx.x;
    if (i < n) p[i] = 0.0f;
}

__global__ void dom_count_kernel(const int* dom, int* domCnt) {
    int i = blockIdx.x * blockDim.x + threadIdx.x;
    if (i < B_N) atomicAdd(&domCnt[dom[i]], 1);
}

__global__ void cvt_f16_kernel(const float* x, _Float16* y, int n) {
    int i = blockIdx.x * blockDim.x + threadIdx.x;
    if (i < n) y[i] = (_Float16)x[i];
}

__global__ void build_aug_kernel(const float* e1, const float* lm, const int* mix, _Float16* augh) {
    int i = blockIdx.x * blockDim.x + threadIdx.x;
    if (i < B_N * D_N) {
        int j = i >> 9;          // / D_N
        int d = i & (D_N - 1);   // % D_N
        float l = lm[j];
        float v = l * e1[i] + (1.0f - l) * e1[mix[j] * D_N + d];
        augh[i] = (_Float16)v;
    }
}

__global__ void transpose_W_kernel(const float* W, _Float16* Wt) {
    int i = blockIdx.x * blockDim.x + threadIdx.x;
    if (i < D_N * D_N) {
        int c = i >> 9;
        int d = i & (D_N - 1);
        Wt[i] = (_Float16)W[d * D_N + c];   // Wt[c][d] = W[d][c]
    }
}

// ---------- phi_aug = aug @ W_e2 + b_e2 (f16 WMMA, f32 accum), stored f16 ----------
__global__ __launch_bounds__(32) void gemm_phi_kernel(const _Float16* __restrict__ augh,
                                                      const _Float16* __restrict__ Wth,
                                                      const float* __restrict__ b_e2,
                                                      _Float16* __restrict__ phi) {
    int ct = blockIdx.x;   // 0..31  (D/16)
    int rt = blockIdx.y;   // 0..255 (B/16)
    int lane = threadIdx.x;
    v8f c = {};
    for (int kc = 0; kc < D_N; kc += 32) {
        v16h a = load_fragA(augh, D_N, rt * 16, kc);
        v16h b = load_fragB(Wth,  D_N, ct * 16, kc);
        c = __builtin_amdgcn_wmma_f32_16x16x32_f16(false, a, false, b, (short)0, c, false, false);
    }
    int col   = ct * 16 + (lane & 15);
    int rbase = rt * 16 + (lane >> 4) * 8;
    float bias = b_e2[col];
#pragma unroll
    for (int v = 0; v < 8; ++v)
        phi[(rbase + v) * D_N + col] = (_Float16)(c[v] + bias);
}

// ---------- L_supp partials: exp(tau * e2@e2.T) row/domain sums via WMMA ----------
// Grid: (NCG, B/16). One wave per block; A fragments hoisted across the col-tile loop.
__global__ __launch_bounds__(32) void supp_partial_kernel(const _Float16* __restrict__ e2h,
                                                          const int* __restrict__ dom,
                                                          float* __restrict__ denArr,
                                                          float* __restrict__ sameArr) {
    int cg   = blockIdx.x;       // column-tile group
    int rt   = blockIdx.y;       // row tile
    int lane = threadIdx.x;
    int halfId = lane >> 4;

    // Hoist the 16 A fragments for this row tile (K = 0..511)
    v16h afr[16];
#pragma unroll
    for (int kk = 0; kk < 16; ++kk) afr[kk] = load_fragA(e2h, D_N, rt * 16, kk * 32);

    float den[8], same[8];
    int domRow[8];
#pragma unroll
    for (int v = 0; v < 8; ++v) {
        den[v] = 0.0f; same[v] = 0.0f;
        domRow[v] = dom[rt * 16 + halfId * 8 + v];
    }

    int ct0 = cg * CT_PER_CG;
    for (int t = 0; t < CT_PER_CG; ++t) {
        int ct = ct0 + t;
        if (t + 1 < CT_PER_CG) {
            // prefetch next column tile's row for this lane (global_prefetch_b8)
            __builtin_prefetch(e2h + (size_t)((ct + 1) * 16 + (lane & 15)) * D_N, 0, 1);
        }
        v8f c = {};
#pragma unroll
        for (int kk = 0; kk < 16; ++kk) {
            v16h b = load_fragB(e2h, D_N, ct * 16, kk * 32);
            c = __builtin_amdgcn_wmma_f32_16x16x32_f16(false, afr[kk], false, b, (short)0, c, false, false);
        }
        int dcol = dom[ct * 16 + (lane & 15)];
#pragma unroll
        for (int v = 0; v < 8; ++v) {
            float e = __expf(TAUF * c[v]);
            den[v] += e;
            if (dcol == domRow[v]) same[v] += e;
        }
    }
    // reduce across the 16 lanes sharing each row (xor of bits 0..3 stays within half)
#pragma unroll
    for (int v = 0; v < 8; ++v) {
        for (int m = 1; m < 16; m <<= 1) {
            den[v]  += __shfl_xor(den[v],  m, 32);
            same[v] += __shfl_xor(same[v], m, 32);
        }
    }
    if ((lane & 15) == 0) {
#pragma unroll
        for (int v = 0; v < 8; ++v) {
            int row = rt * 16 + halfId * 8 + v;
            atomicAdd(&denArr[row],  den[v]);
            atomicAdd(&sameArr[row], same[v]);
        }
    }
}

__global__ void supp_final_kernel(const float* denArr, const float* sameArr,
                                  const int* dom, const int* domCnt, float* acc) {
    int j = blockIdx.x * blockDim.x + threadIdx.x;
    if (j < B_N) {
        float d   = denArr[j];
        float nom = d - sameArr[j];       // diagonal is same-domain -> excluded, matches diff_dom
        float Ls  = __logf(nom / (d + EPSF));
        bool anyDiff = (B_N - domCnt[dom[j]]) > 0;
        bool valid   = anyDiff && (Ls == Ls);
        if (valid) atomicAdd(&acc[6], Ls);
    }
}

// ---------- L_aug: per-row pos/neg gathered dot products ----------
__global__ __launch_bounds__(128) void aug_loss_kernel(const float* __restrict__ e2,
                                                       const _Float16* __restrict__ phi,
                                                       const int* __restrict__ neg_idx,
                                                       float* acc) {
    int j    = blockIdx.x;
    int tid  = threadIdx.x;
    int lane = tid & 31;
    int wave = tid >> 5;
    __shared__ float se2[D_N];
    __shared__ float s_pos, s_neg;
    if (tid == 0) { s_pos = 0.0f; s_neg = 0.0f; }
    for (int d = tid; d < D_N; d += 128) se2[d] = e2[j * D_N + d];
    __syncthreads();
    for (int t = wave; t < NNEG + 1; t += 4) {
        int r = (t == 0) ? j : neg_idx[j * NNEG + (t - 1)];
        const _Float16* pr = phi + (size_t)r * D_N;
        float p = 0.0f;
        for (int d = lane; d < D_N; d += 32) p += se2[d] * (float)pr[d];
#pragma unroll
        for (int m = 16; m >= 1; m >>= 1) p += __shfl_xor(p, m, 32);
        if (lane == 0) {
            float e = __expf(TAUF * p);
            if (t == 0) s_pos = e; else atomicAdd(&s_neg, e);
        }
    }
    __syncthreads();
    if (tid == 0) {
        float pos = s_pos;
        float La  = __logf(pos / (EPSF + pos + s_neg));
        if (La != La) La = 0.0f;   // NaN -> 0 per reference
        atomicAdd(&acc[5], La);
    }
}

// ---------- mse / sum-of-squares / finalize ----------
__global__ void mse_kernel(const float* yp, const float* yt, float* acc) {
    int i = blockIdx.x * blockDim.x + threadIdx.x;
    if (i < B_N) {
        float d = yp[i] - yt[i];
        atomicAdd(&acc[0], d * d);
    }
}

__global__ void sumsq_kernel(const float* x, int n, float* slot) {
    __shared__ float sh[256];
    float s = 0.0f;
    for (int i = blockIdx.x * blockDim.x + threadIdx.x; i < n; i += gridDim.x * blockDim.x)
        s += x[i] * x[i];
    sh[threadIdx.x] = s;
    __syncthreads();
    for (int o = 128; o > 0; o >>= 1) {
        if (threadIdx.x < o) sh[threadIdx.x] += sh[threadIdx.x + o];
        __syncthreads();
    }
    if (threadIdx.x == 0) atomicAdd(slot, sh[0]);
}

__global__ void finalize_kernel(const float* acc, float* out) {
    if (threadIdx.x == 0) {
        out[0] = acc[0] / (float)B_N;
        out[1] = REGW * (sqrtf(acc[1]) + sqrtf(acc[2]) + sqrtf(acc[3]) + sqrtf(acc[4]));
        out[2] = -acc[5] / (float)B_N;
        out[3] = -acc[6] / (float)B_N;
    }
}

extern "C" void kernel_launch(void* const* d_in, const int* in_sizes, int n_in,
                              void* d_out, int out_size, void* d_ws, size_t ws_size,
                              hipStream_t stream) {
    const float* e1      = (const float*)d_in[0];
    const float* e2      = (const float*)d_in[1];
    const float* y_pred  = (const float*)d_in[2];
    const float* y_targ  = (const float*)d_in[3];
    const float* W_e2    = (const float*)d_in[4];
    const float* b_e2    = (const float*)d_in[5];
    const float* W_head  = (const float*)d_in[6];
    const float* b_head  = (const float*)d_in[7];
    const float* lmbda   = (const float*)d_in[8];
    const int*   mix_idx = (const int*)d_in[9];
    const int*   neg_idx = (const int*)d_in[10];
    const int*   dom_tag = (const int*)d_in[11];
    float* out = (float*)d_out;

    char* base = (char*)d_ws;
    float* acc    = (float*)base;                 // 8 floats
    int*   domCnt = (int*)(base + 32);            // 8 ints
    _Float16* e2h  = (_Float16*)(base + 256);     // [B,D]
    _Float16* augh = e2h  + (size_t)B_N * D_N;    // [B,D]
    _Float16* Wth  = augh + (size_t)B_N * D_N;    // [D,D] transposed
    _Float16* phi  = Wth  + (size_t)D_N * D_N;    // [B,D] phi_aug (f16)
    float* denArr  = (float*)(phi + (size_t)B_N * D_N);  // [B]
    float* sameArr = denArr + B_N;                       // [B]
    (void)in_sizes; (void)n_in; (void)out_size; (void)ws_size;

    const int BD = B_N * D_N;

    init_ws_kernel<<<1, 32, 0, stream>>>(acc, domCnt);
    zero_f32_kernel<<<(2 * B_N + 255) / 256, 256, 0, stream>>>(denArr, 2 * B_N);
    dom_count_kernel<<<(B_N + 255) / 256, 256, 0, stream>>>(dom_tag, domCnt);

    cvt_f16_kernel<<<(BD + 255) / 256, 256, 0, stream>>>(e2, e2h, BD);
    build_aug_kernel<<<(BD + 255) / 256, 256, 0, stream>>>(e1, lmbda, mix_idx, augh);
    transpose_W_kernel<<<(D_N * D_N + 255) / 256, 256, 0, stream>>>(W_e2, Wth);

    gemm_phi_kernel<<<dim3(D_N / 16, B_N / 16), 32, 0, stream>>>(augh, Wth, b_e2, phi);

    supp_partial_kernel<<<dim3(NCG, B_N / 16), 32, 0, stream>>>(e2h, dom_tag, denArr, sameArr);
    supp_final_kernel<<<(B_N + 255) / 256, 256, 0, stream>>>(denArr, sameArr, dom_tag, domCnt, acc);

    aug_loss_kernel<<<B_N, 128, 0, stream>>>(e2, phi, neg_idx, acc);

    mse_kernel<<<(B_N + 255) / 256, 256, 0, stream>>>(y_pred, y_targ, acc);
    sumsq_kernel<<<256, 256, 0, stream>>>(W_e2,   D_N * D_N, &acc[1]);
    sumsq_kernel<<<2,   256, 0, stream>>>(b_e2,   D_N,       &acc[2]);
    sumsq_kernel<<<2,   256, 0, stream>>>(W_head, D_N,       &acc[3]);
    sumsq_kernel<<<1,   256, 0, stream>>>(b_head, 1,         &acc[4]);

    finalize_kernel<<<1, 32, 0, stream>>>(acc, out);
}
